// RegularizerHead_71339406786805
// MI455X (gfx1250) — matrix operand order repaired and checked by reading, not compiled, not run on hardware
//
#include <hip/hip_runtime.h>
#include <math.h>

#define G_NUM 1024
#define NPG   1024
#define HID   128
#define WROW  136   // padded bf16 row stride (272 B): bank-conflict-free B frags

typedef __attribute__((ext_vector_type(16))) __bf16 v16bf;
typedef __attribute__((ext_vector_type(8)))  __bf16 v8bf;
typedef __attribute__((ext_vector_type(4)))  __bf16 v4bf;
typedef __attribute__((ext_vector_type(8)))  float  v8f;
typedef __attribute__((ext_vector_type(4)))  float  v4f;

// ---- LDS layout (bytes) ----
#define OFF_WTD   0        // d_w1^T bf16 [128][136]  34816
#define OFF_WTU   34816    // u_w1^T bf16             34816
#define OFF_W2TD  69632    // d_w2^T bf16 [16][136]    4352 (rows 2..15 zero)
#define OFF_W2TU  73984    // u_w2^T bf16              4352
#define OFF_HB    78336    // H chunk / t bf16 [128][136] 34816
#define OFF_B1D   113152   // f32[128]
#define OFF_B1U   113664
#define OFF_B2D   114176   // f32[16]
#define OFF_B2U   114240
#define OFF_PPART 114304   // f32[8][128] pooling partials
#define OFF_POOL  118400   // f32[128]
#define OFF_SHID  118912   // f32[128]
#define SMEM_BYTES 119424

static __device__ __forceinline__ float softplusf(float x) {
  return (x > 20.0f) ? x : log1pf(__expf(x));
}

// Build a 16-element bf16 fragment from two 16-byte LDS loads.
static __device__ __forceinline__ v16bf ldfrag(const __bf16* p0, const __bf16* p1) {
  v8bf lo = *(const v8bf*)p0;
  v8bf hi = *(const v8bf*)p1;
  v16bf r;
#pragma unroll
  for (int i = 0; i < 8; ++i) { r[i] = lo[i]; r[i + 8] = hi[i]; }
  return r;
}

static __device__ __forceinline__ v8f wmma_bf16(v16bf a, v16bf b, v8f c) {
  return __builtin_amdgcn_wmma_f32_16x16x32_bf16(false, a, false, b, (short)0, c,
                                                 false, false);
}

__global__ __launch_bounds__(256, 1)
void reg_head_fused(const float* __restrict__ H,
                    const float* __restrict__ d_w1, const float* __restrict__ d_b1,
                    const float* __restrict__ d_w2, const float* __restrict__ d_b2,
                    const float* __restrict__ u_w1, const float* __restrict__ u_b1,
                    const float* __restrict__ u_w2, const float* __restrict__ u_b2,
                    const float* __restrict__ s_w1, const float* __restrict__ s_b1,
                    const float* __restrict__ s_w2, const float* __restrict__ s_b2,
                    float* __restrict__ dOut, float* __restrict__ UOut,
                    float* __restrict__ sOut) {
  extern __shared__ __align__(16) unsigned char smem[];
  __bf16* wTd  = (__bf16*)(smem + OFF_WTD);
  __bf16* wTu  = (__bf16*)(smem + OFF_WTU);
  __bf16* w2Td = (__bf16*)(smem + OFF_W2TD);
  __bf16* w2Tu = (__bf16*)(smem + OFF_W2TU);
  __bf16* Hb   = (__bf16*)(smem + OFF_HB);
  float* b1d   = (float*)(smem + OFF_B1D);
  float* b1u   = (float*)(smem + OFF_B1U);
  float* b2d   = (float*)(smem + OFF_B2D);
  float* b2u   = (float*)(smem + OFF_B2U);
  float* ppart = (float*)(smem + OFF_PPART);
  float* pool  = (float*)(smem + OFF_POOL);
  float* shid  = (float*)(smem + OFF_SHID);

  const int g    = blockIdx.x;
  const int tid  = threadIdx.x;
  const int wave = tid >> 5;
  const int lane = tid & 31;
  const int half = lane >> 4;   // 0: lanes 0-15, 1: lanes 16-31
  const int l16  = lane & 15;

  // ---- stage weights: transpose + convert to bf16 (L2-resident reads) ----
  for (int idx = tid; idx < HID * HID; idx += 256) {
    int k = idx >> 7, n = idx & 127;
    wTd[n * WROW + k] = (__bf16)d_w1[idx];
    wTu[n * WROW + k] = (__bf16)u_w1[idx];
  }
  for (int idx = tid; idx < 16 * HID; idx += 256) {
    int n = idx >> 7, k = idx & 127;
    w2Td[n * WROW + k] = (n < 2) ? (__bf16)d_w2[k * 2 + n] : (__bf16)0.0f;
    w2Tu[n * WROW + k] = (__bf16)u_w2[k * 16 + n];
  }
  if (tid < HID) { b1d[tid] = d_b1[tid]; b1u[tid] = u_b1[tid]; }
  if (tid < 16)  { b2d[tid] = (tid < 2) ? d_b2[tid] : 0.0f; b2u[tid] = u_b2[tid]; }
  __syncthreads();

  const v8f zeros = {0.f, 0.f, 0.f, 0.f, 0.f, 0.f, 0.f, 0.f};
  float pax = 0.f, pay = 0.f, paz = 0.f, paw = 0.f;  // pool acc, cols lane*4..+3
  const int m0 = wave * 16;                           // wave-private LDS rows

  // hoist loop-invariant bias values out of the chunk loop
  float bn_d[8], bn_u[8];
#pragma unroll
  for (int t = 0; t < 8; ++t) {
    bn_d[t] = b1d[t * 16 + l16];
    bn_u[t] = b1u[t * 16 + l16];
  }
  const float b2d_l = b2d[l16];
  const float b2u_l = b2u[l16];

  const __bf16* arow = Hb + (m0 + l16) * WROW;  // A frag row for this lane

  // ---- main loop: 8 chunks of 128 rows, wave owns a 16-row strip ----
  for (int ch = 0; ch < 8; ++ch) {
    const int rowbase = g * NPG + ch * 128 + wave * 16;
    // per-chunk base pointers: all inner accesses use small immediate offsets
    const float* hbase = H + (size_t)rowbase * HID + (unsigned)(lane * 4);
    float*       ubase = UOut + (size_t)rowbase * 16 + (unsigned)l16;
    float*       dbase = dOut + (size_t)rowbase * 2 + (unsigned)l16;

    // stage strip f32 -> bf16 (non-temporal: H is streamed once), pool in f32
#pragma unroll
    for (int rr = 0; rr < 16; ++rr) {
      const v4f v = __builtin_nontemporal_load((const v4f*)(hbase + rr * HID));
      pax += v.x; pay += v.y; paz += v.z; paw += v.w;
      v4bf p;
      p[0] = (__bf16)v.x; p[1] = (__bf16)v.y; p[2] = (__bf16)v.z; p[3] = (__bf16)v.w;
      *(v4bf*)(Hb + (m0 + rr) * WROW + lane * 4) = p;
    }

    // ---- layer 1, both heads: 64 WMMAs per strip ----
    v8f accd[8], accu[8];
#pragma unroll
    for (int t = 0; t < 8; ++t) { accd[t] = zeros; accu[t] = zeros; }

#pragma unroll
    for (int c = 0; c < 4; ++c) {
      const __bf16* ap = arow + c * 32 + half * 16;
      v16bf a = ldfrag(ap, ap + 8);
#pragma unroll
      for (int t = 0; t < 8; ++t) {
        const __bf16* bp = wTd + (t * 16 + l16) * WROW + c * 32 + half * 16;
        accd[t] = wmma_bf16(a, ldfrag(bp, bp + 8), accd[t]);
        const __bf16* bq = wTu + (t * 16 + l16) * WROW + c * 32 + half * 16;
        accu[t] = wmma_bf16(a, ldfrag(bq, bq + 8), accu[t]);
      }
    }

    // ---- head d: bias+relu -> LDS(bf16) -> layer-2 WMMA -> softplus -> store ----
#pragma unroll
    for (int t = 0; t < 8; ++t) {
#pragma unroll
      for (int r = 0; r < 8; ++r) {
        float x = fmaxf(accd[t][r] + bn_d[t], 0.0f);
        Hb[(m0 + r + 8 * half) * WROW + t * 16 + l16] = (__bf16)x;
      }
    }
    v8f od = zeros;
#pragma unroll
    for (int c = 0; c < 4; ++c) {
      const __bf16* ap = arow + c * 32 + half * 16;
      const __bf16* bp = w2Td + l16 * WROW + c * 32 + half * 16;
      od = wmma_bf16(ldfrag(ap, ap + 8), ldfrag(bp, bp + 8), od);
    }
    if (l16 < 2) {
#pragma unroll
      for (int r = 0; r < 8; ++r) {
        __builtin_nontemporal_store(softplusf(od[r] + b2d_l),
                                    dbase + (r + 8 * half) * 2);
      }
    }

    // ---- head u: bias+relu -> LDS(bf16) -> layer-2 WMMA -> store (no softplus) ----
#pragma unroll
    for (int t = 0; t < 8; ++t) {
#pragma unroll
      for (int r = 0; r < 8; ++r) {
        float x = fmaxf(accu[t][r] + bn_u[t], 0.0f);
        Hb[(m0 + r + 8 * half) * WROW + t * 16 + l16] = (__bf16)x;
      }
    }
    v8f ou = zeros;
#pragma unroll
    for (int c = 0; c < 4; ++c) {
      const __bf16* ap = arow + c * 32 + half * 16;
      const __bf16* bp = w2Tu + l16 * WROW + c * 32 + half * 16;
      ou = wmma_bf16(ldfrag(ap, ap + 8), ldfrag(bp, bp + 8), ou);
    }
#pragma unroll
    for (int r = 0; r < 8; ++r) {
      __builtin_nontemporal_store(ou[r] + b2u_l,
                                  ubase + (r + 8 * half) * 16);  // 64B segments
    }
  }

  // ---- pooling: deterministic tree reduction, then per-graph s-MLP ----
  ppart[wave * 128 + lane * 4 + 0] = pax;
  ppart[wave * 128 + lane * 4 + 1] = pay;
  ppart[wave * 128 + lane * 4 + 2] = paz;
  ppart[wave * 128 + lane * 4 + 3] = paw;
  __syncthreads();
  if (tid < 128) {
    float s = 0.f;
#pragma unroll
    for (int w = 0; w < 8; ++w) s += ppart[w * 128 + tid];
    pool[tid] = s * (1.0f / (float)NPG);  // mean
  }
  __syncthreads();
  if (tid < 128) {
    float acc = s_b1[tid];
#pragma unroll 8
    for (int k = 0; k < 128; ++k) acc += pool[k] * s_w1[k * 128 + tid];
    shid[tid] = fmaxf(acc, 0.0f);
  }
  __syncthreads();
  if (tid < 8) {
    float acc = s_b2[tid];
#pragma unroll 8
    for (int k = 0; k < 128; ++k) acc += shid[k] * s_w2[k * 8 + tid];
    sOut[(size_t)g * 8 + tid] = softplusf(acc);
  }
}

extern "C" void kernel_launch(void* const* d_in, const int* in_sizes, int n_in,
                              void* d_out, int out_size, void* d_ws, size_t ws_size,
                              hipStream_t stream) {
  (void)in_sizes; (void)n_in; (void)out_size; (void)d_ws; (void)ws_size;
  const float* H    = (const float*)d_in[0];
  // d_in[1] = batch (int32) unused: graphs are equal-sized, WG g == graph g
  const float* d_w1 = (const float*)d_in[2];
  const float* d_b1 = (const float*)d_in[3];
  const float* d_w2 = (const float*)d_in[4];
  const float* d_b2 = (const float*)d_in[5];
  const float* u_w1 = (const float*)d_in[6];
  const float* u_b1 = (const float*)d_in[7];
  const float* u_w2 = (const float*)d_in[8];
  const float* u_b2 = (const float*)d_in[9];
  const float* s_w1 = (const float*)d_in[10];
  const float* s_b1 = (const float*)d_in[11];
  const float* s_w2 = (const float*)d_in[12];
  const float* s_b2 = (const float*)d_in[13];

  float* out = (float*)d_out;
  float* dO = out;                                  // [1024, 2048]
  float* UO = out + (size_t)G_NUM * NPG * 2;        // [1024, 2048, 8]
  float* sO = UO + (size_t)G_NUM * NPG * 16;        // [1024, 8]

  hipFuncSetAttribute((const void*)reg_head_fused,
                      hipFuncAttributeMaxDynamicSharedMemorySize, SMEM_BYTES);
  reg_head_fused<<<dim3(G_NUM), dim3(256), SMEM_BYTES, stream>>>(
      H, d_w1, d_b1, d_w2, d_b2, u_w1, u_b1, u_w2, u_b2,
      s_w1, s_b1, s_w2, s_b2, dO, UO, sO);
}
